// TranscendentNeuralLayer_50964081934960
// MI455X (gfx1250) — matrix operand order repaired and checked by reading, not compiled, not run on hardware
//
#include <hip/hip_runtime.h>
#include <hip/hip_bf16.h>

// ---------------------------------------------------------------------------
// out = tanh(x @ Wc + bc + prob),  prob[j] = (1/n)*(prod_{d,g} cos(tw[d,j,g]))^2
// (quantum scan collapses: column-constant state => per-row cos products)
// Heavy part: 4096x2048x2048 f32 GEMM -> bf16 hi/lo split (3 WMMAs, ~fp32 acc).
// v3: WG tile 128x128, wave tile 64x32 (better WMMA:LDS ratio, half the x
//     L2 re-reads); async GLOBAL_LOAD_ASYNC_TO_LDS_B128 ping-pong staging.
// ---------------------------------------------------------------------------

#define BDIM   4096
#define KDIM   2048
#define NDIM   2048
#define DEPTH  7

#define KT     32
// async kernel tiles
#define TM_A   128
#define TN_A   128
#define LDA_A  40          // LDS row stride (halfwords): 80B rows -> 16B-aligned units
// fallback kernel tiles
#define TM_F   128
#define TN_F   64
#define LDA_F  34

typedef __bf16 v16bf __attribute__((ext_vector_type(16)));
typedef float  v8f   __attribute__((ext_vector_type(8)));

__device__ __forceinline__ unsigned short f32_to_bf16_rne(float f) {
    unsigned int u = __float_as_uint(f);
    unsigned int r = u + 0x7FFFu + ((u >> 16) & 1u);
    return (unsigned short)(r >> 16);
}
__device__ __forceinline__ float bf16_bits_to_f32(unsigned short h) {
    return __uint_as_float(((unsigned int)h) << 16);
}
__device__ __forceinline__ void split_bf16(float v, unsigned short& h, unsigned short& l) {
    h = f32_to_bf16_rne(v);
    l = f32_to_bf16_rne(v - bf16_bits_to_f32(h));
}

// ---- CDNA5 async global->LDS copy (ASYNCcnt path, CDNA5 ISA) ------------
__device__ __forceinline__ void async_copy_b128(unsigned lds_byte_addr, const void* gaddr) {
    asm volatile("global_load_async_to_lds_b128 %0, %1, off"
                 :: "v"(lds_byte_addr), "v"(gaddr)
                 : "memory");
}
__device__ __forceinline__ void wait_asynccnt0() {
    asm volatile("s_wait_asynccnt 0x0" ::: "memory");
}
__device__ __forceinline__ unsigned lds_addr_of(const void* p) {
    return (unsigned)(size_t)p;   // low 32 bits of shared-aperture pointer = LDS offset
}

// ---------------------------------------------------------------------------
// Kernel 1: addvec[j] = bc[j] + (1/n) * (prod cos)^2
// ---------------------------------------------------------------------------
__global__ __launch_bounds__(256) void tnl_prob_kernel(
    const float* __restrict__ tw, const float* __restrict__ bc,
    float* __restrict__ addvec) {
    int j = blockIdx.x * blockDim.x + threadIdx.x;
    if (j >= NDIM) return;
    float p = 1.0f;
#pragma unroll
    for (int d = 0; d < DEPTH; ++d) {
        const float* w = tw + ((size_t)d * KDIM + (size_t)j) * (size_t)NDIM;
        p *= cosf(w[0]);
        p *= cosf(w[1]);
        p *= cosf(w[2]);
    }
    addvec[j] = bc[j] + p * p * (1.0f / (float)KDIM);
}

// ---------------------------------------------------------------------------
// Kernel 2a: x -> bf16 hi/lo planes (row-major, [m][k])
// ---------------------------------------------------------------------------
__global__ __launch_bounds__(256) void tnl_cvt_x_kernel(
    const float* __restrict__ x,
    unsigned short* __restrict__ xhi, unsigned short* __restrict__ xlo) {
    size_t base = ((size_t)blockIdx.x * 256 + threadIdx.x) * 8;
    unsigned short h[8], l[8];
#pragma unroll
    for (int i = 0; i < 8; i += 4) {
        float4 t = *(const float4*)(x + base + i);
        split_bf16(t.x, h[i + 0], l[i + 0]);
        split_bf16(t.y, h[i + 1], l[i + 1]);
        split_bf16(t.z, h[i + 2], l[i + 2]);
        split_bf16(t.w, h[i + 3], l[i + 3]);
    }
    *(uint4*)(xhi + base) = *(const uint4*)h;
    *(uint4*)(xlo + base) = *(const uint4*)l;
}

// ---------------------------------------------------------------------------
// Kernel 2b: W[k][n] -> transposed bf16 hi/lo planes [n][k] (LDS 32x32 tiles)
// ---------------------------------------------------------------------------
__global__ __launch_bounds__(256) void tnl_cvt_w_kernel(
    const float* __restrict__ w,
    unsigned short* __restrict__ whit, unsigned short* __restrict__ wlot) {
    __shared__ unsigned short sHi[32][33];
    __shared__ unsigned short sLo[32][33];
    const int tid = threadIdx.x;
    const int k0 = blockIdx.x * 32;
    const int n0 = blockIdx.y * 32;
#pragma unroll
    for (int rep = 0; rep < 4; ++rep) {
        int kr = (tid >> 5) + rep * 8;
        int nc = tid & 31;
        float v = w[(size_t)(k0 + kr) * NDIM + (n0 + nc)];
        unsigned short h, l;
        split_bf16(v, h, l);
        sHi[kr][nc] = h;
        sLo[kr][nc] = l;
    }
    __syncthreads();
#pragma unroll
    for (int rep = 0; rep < 2; ++rep) {
        int n  = (tid >> 4) + rep * 16;
        int kc = (tid & 15) * 2;
        unsigned uh = (unsigned)sHi[kc][n] | ((unsigned)sHi[kc + 1][n] << 16);
        unsigned ul = (unsigned)sLo[kc][n] | ((unsigned)sLo[kc + 1][n] << 16);
        *(unsigned*)(whit + (size_t)(n0 + n) * KDIM + k0 + kc) = uh;
        *(unsigned*)(wlot + (size_t)(n0 + n) * KDIM + k0 + kc) = ul;
    }
}

// ---------------------------------------------------------------------------
// Kernel 3: WMMA GEMM from pre-converted bf16 planes.
//   128x128 WG tile, 8 waves (2 along M x 4 along N), wave tile 64x32.
//   Async double-buffered LDS staging (ping-pong).
// ---------------------------------------------------------------------------
__global__ __launch_bounds__(256) void tnl_gemm_async_kernel(
    const unsigned short* __restrict__ xhi, const unsigned short* __restrict__ xlo,
    const unsigned short* __restrict__ whit, const unsigned short* __restrict__ wlot,
    const float* __restrict__ addvec, float* __restrict__ out) {

    __shared__ __align__(16) unsigned short sA[2][2][TM_A * LDA_A]; // [buf][hi/lo]
    __shared__ __align__(16) unsigned short sB[2][2][TN_A * LDA_A];

    const int tid  = threadIdx.x;
    const int lane = tid & 31;
    const int wave = tid >> 5;
    const int waveM = wave & 1;   // 0..1 -> 64-row band
    const int waveN = wave >> 1;  // 0..3 -> 32-col band
    const int lsub = lane & 15;
    const int half = lane >> 4;

    const int tileM0 = blockIdx.y * TM_A;
    const int tileN0 = blockIdx.x * TN_A;

    // staging: each plane is 128 rows x 64B = 512 16B-units; 2 units/thread/plane
    const int srow = tid >> 2;   // 0..63, +64 for second unit
    const int sq   = tid & 3;

    auto stage = [&](int buf, int kk) {
        const size_t kbase = (size_t)kk * KT;
#pragma unroll
        for (int j = 0; j < 2; ++j) {
            const int r = srow + j * 64;
            const unsigned loff = (unsigned)(r * (LDA_A * 2) + sq * 16);
            const size_t ga = (size_t)(tileM0 + r) * KDIM + kbase + sq * 8;
            async_copy_b128(lds_addr_of(&sA[buf][0][0]) + loff, xhi + ga);
            async_copy_b128(lds_addr_of(&sA[buf][1][0]) + loff, xlo + ga);
            const size_t gb = (size_t)(tileN0 + r) * KDIM + kbase + sq * 8;
            async_copy_b128(lds_addr_of(&sB[buf][0][0]) + loff, whit + gb);
            async_copy_b128(lds_addr_of(&sB[buf][1][0]) + loff, wlot + gb);
        }
    };

    v8f acc[4][2];
#pragma unroll
    for (int tm = 0; tm < 4; ++tm)
#pragma unroll
        for (int tn = 0; tn < 2; ++tn)
            acc[tm][tn] = (v8f)(0.0f);

    stage(0, 0);

    union Frag { v16bf bf; unsigned int u[8]; };

    const int nk = KDIM / KT;
    for (int kk = 0; kk < nk; ++kk) {
        const int buf = kk & 1;
        wait_asynccnt0();
        __syncthreads();
        if (kk + 1 < nk) stage(buf ^ 1, kk + 1);

        // ---- B fragments (kept live across all 4 M-subtiles) ----
        Frag bhi[2], blo[2];
#pragma unroll
        for (int tn = 0; tn < 2; ++tn) {
            const int n = waveN * 32 + tn * 16 + lsub;
            const unsigned short* hiRow = &sB[buf][0][0] + n * LDA_A;
            const unsigned short* loRow = &sB[buf][1][0] + n * LDA_A;
#pragma unroll
            for (int v = 0; v < 8; ++v) {
                int k = half * 16 + v * 2;   // B 32x16: lanes0-15 K=0..15, lanes16-31 K=16..31
                bhi[tn].u[v] = *(const unsigned int*)(hiRow + k);
                blo[tn].u[v] = *(const unsigned int*)(loRow + k);
            }
        }
        // ---- per-M-subtile: one A hi/lo pair live at a time ----
#pragma unroll
        for (int tm = 0; tm < 4; ++tm) {
            const int m = waveM * 64 + tm * 16 + lsub;
            const unsigned short* hiRow = &sA[buf][0][0] + m * LDA_A;
            const unsigned short* loRow = &sA[buf][1][0] + m * LDA_A;
            Frag ahi, alo;
#pragma unroll
            for (int v = 0; v < 8; ++v) {
                int k = half * 16 + v * 2;  // A 16x32: lanes0-15 K=0..15, lanes16-31 K=16..31
                ahi.u[v] = *(const unsigned int*)(hiRow + k);
                alo.u[v] = *(const unsigned int*)(loRow + k);
            }
#pragma unroll
            for (int tn = 0; tn < 2; ++tn) {
                acc[tm][tn] = __builtin_amdgcn_wmma_f32_16x16x32_bf16(
                    false, ahi.bf, false, bhi[tn].bf, (short)0, acc[tm][tn], false, false);
                acc[tm][tn] = __builtin_amdgcn_wmma_f32_16x16x32_bf16(
                    false, ahi.bf, false, blo[tn].bf, (short)0, acc[tm][tn], false, false);
                acc[tm][tn] = __builtin_amdgcn_wmma_f32_16x16x32_bf16(
                    false, alo.bf, false, bhi[tn].bf, (short)0, acc[tm][tn], false, false);
            }
        }
    }

    // ---- fused epilogue: out = tanh(acc + bias + prob) ----
#pragma unroll
    for (int tn = 0; tn < 2; ++tn) {
        const int gn = tileN0 + waveN * 32 + tn * 16 + lsub;
        const float add = addvec[gn];
#pragma unroll
        for (int tm = 0; tm < 4; ++tm) {
            const int gmBase = tileM0 + waveM * 64 + tm * 16 + half * 8;
#pragma unroll
            for (int v = 0; v < 8; ++v)
                out[(size_t)(gmBase + v) * NDIM + gn] = tanhf(acc[tm][tn][v] + add);
        }
    }
}

// ---------------------------------------------------------------------------
// Fallback GEMM (converts in-loop; needs only 8KB ws)
// ---------------------------------------------------------------------------
__global__ __launch_bounds__(256) void tnl_gemm_fallback_kernel(
    const float* __restrict__ x, const float* __restrict__ Wc,
    const float* __restrict__ addvec, float* __restrict__ out) {

    __shared__ __align__(16) unsigned short sAhi[TM_F * LDA_F];
    __shared__ __align__(16) unsigned short sAlo[TM_F * LDA_F];
    __shared__ __align__(16) unsigned short sBhi[TN_F * LDA_F];
    __shared__ __align__(16) unsigned short sBlo[TN_F * LDA_F];

    const int tid  = threadIdx.x;
    const int lane = tid & 31;
    const int wave = tid >> 5;
    const int waveM = wave & 3;
    const int waveN = wave >> 2;
    const int lsub = lane & 15;
    const int half = lane >> 4;

    const int tileM0 = blockIdx.y * TM_F;
    const int tileN0 = blockIdx.x * TN_F;

    const int ar = tid >> 1;
    const int ac = (tid & 1) * 16;
    const int bk = tid >> 3;
    const int bn = (tid & 7) * 8;

    const float* aRow = x  + (size_t)(tileM0 + ar) * KDIM + ac;
    const float* bRow = Wc + (size_t)bk * NDIM + tileN0 + bn;

    float aReg[16];
    float bReg[8];
#pragma unroll
    for (int i = 0; i < 16; i += 4) {
        float4 t = *(const float4*)(aRow + i);
        aReg[i] = t.x; aReg[i + 1] = t.y; aReg[i + 2] = t.z; aReg[i + 3] = t.w;
    }
#pragma unroll
    for (int i = 0; i < 8; i += 4) {
        float4 t = *(const float4*)(bRow + i);
        bReg[i] = t.x; bReg[i + 1] = t.y; bReg[i + 2] = t.z; bReg[i + 3] = t.w;
    }

    v8f acc[2][2];
#pragma unroll
    for (int tm = 0; tm < 2; ++tm)
#pragma unroll
        for (int tn = 0; tn < 2; ++tn)
            acc[tm][tn] = (v8f)(0.0f);

    const int nk = KDIM / KT;
    for (int kk = 0; kk < nk; ++kk) {
        __syncthreads();
        {
            unsigned short* hiRow = sAhi + ar * LDA_F + ac;
            unsigned short* loRow = sAlo + ar * LDA_F + ac;
#pragma unroll
            for (int i = 0; i < 16; i += 2) {
                unsigned short h0, l0, h1, l1;
                split_bf16(aReg[i], h0, l0);
                split_bf16(aReg[i + 1], h1, l1);
                *(unsigned int*)(hiRow + i) = (unsigned int)h0 | ((unsigned int)h1 << 16);
                *(unsigned int*)(loRow + i) = (unsigned int)l0 | ((unsigned int)l1 << 16);
            }
        }
#pragma unroll
        for (int i = 0; i < 8; ++i) {
            unsigned short h, l;
            split_bf16(bReg[i], h, l);
            sBhi[(bn + i) * LDA_F + bk] = h;
            sBlo[(bn + i) * LDA_F + bk] = l;
        }
        __syncthreads();

        if (kk + 1 < nk) {
            const float* aN = aRow + (kk + 1) * KT;
            const float* bN = bRow + (size_t)(kk + 1) * KT * NDIM;
#pragma unroll
            for (int i = 0; i < 16; i += 4) {
                float4 t = *(const float4*)(aN + i);
                aReg[i] = t.x; aReg[i + 1] = t.y; aReg[i + 2] = t.z; aReg[i + 3] = t.w;
            }
#pragma unroll
            for (int i = 0; i < 8; i += 4) {
                float4 t = *(const float4*)(bN + i);
                bReg[i] = t.x; bReg[i + 1] = t.y; bReg[i + 2] = t.z; bReg[i + 3] = t.w;
            }
        }

        union Frag { v16bf bf; unsigned int u[8]; };
        Frag ahi[2], alo[2], bhi[2], blo[2];
#pragma unroll
        for (int tm = 0; tm < 2; ++tm) {
            const int m = waveM * 32 + tm * 16 + lsub;
            const unsigned short* hiRow = sAhi + m * LDA_F;
            const unsigned short* loRow = sAlo + m * LDA_F;
#pragma unroll
            for (int v = 0; v < 8; ++v) {
                int k = half * 16 + v * 2;
                ahi[tm].u[v] = *(const unsigned int*)(hiRow + k);
                alo[tm].u[v] = *(const unsigned int*)(loRow + k);
            }
        }
#pragma unroll
        for (int tn = 0; tn < 2; ++tn) {
            const int n = waveN * 32 + tn * 16 + lsub;
            const unsigned short* hiRow = sBhi + n * LDA_F;
            const unsigned short* loRow = sBlo + n * LDA_F;
#pragma unroll
            for (int v = 0; v < 8; ++v) {
                int k = half * 16 + v * 2;
                bhi[tn].u[v] = *(const unsigned int*)(hiRow + k);
                blo[tn].u[v] = *(const unsigned int*)(loRow + k);
            }
        }
#pragma unroll
        for (int tm = 0; tm < 2; ++tm)
#pragma unroll
            for (int tn = 0; tn < 2; ++tn) {
                acc[tm][tn] = __builtin_amdgcn_wmma_f32_16x16x32_bf16(
                    false, ahi[tm].bf, false, bhi[tn].bf, (short)0, acc[tm][tn], false, false);
                acc[tm][tn] = __builtin_amdgcn_wmma_f32_16x16x32_bf16(
                    false, ahi[tm].bf, false, blo[tn].bf, (short)0, acc[tm][tn], false, false);
                acc[tm][tn] = __builtin_amdgcn_wmma_f32_16x16x32_bf16(
                    false, alo[tm].bf, false, bhi[tn].bf, (short)0, acc[tm][tn], false, false);
            }
    }

#pragma unroll
    for (int tn = 0; tn < 2; ++tn) {
        const int gn = tileN0 + waveN * 32 + tn * 16 + lsub;
        const float add = addvec[gn];
#pragma unroll
        for (int tm = 0; tm < 2; ++tm) {
            const int gmBase = tileM0 + waveM * 32 + tm * 16 + half * 8;
#pragma unroll
            for (int v = 0; v < 8; ++v)
                out[(size_t)(gmBase + v) * NDIM + gn] = tanhf(acc[tm][tn][v] + add);
        }
    }
}

// ---------------------------------------------------------------------------
extern "C" void kernel_launch(void* const* d_in, const int* in_sizes, int n_in,
                              void* d_out, int out_size, void* d_ws, size_t ws_size,
                              hipStream_t stream) {
    const float* x  = (const float*)d_in[0];  // [4096, 2048]
    const float* tw = (const float*)d_in[1];  // [7, 2048, 2048]
    // d_in[2] transcendent_biases: unused by the reference
    const float* cw = (const float*)d_in[3];  // [2048, 2048]
    const float* cb = (const float*)d_in[4];  // [2048]
    float* out = (float*)d_out;
    float* addvec = (float*)d_ws;             // 2048 floats

    tnl_prob_kernel<<<NDIM / 256, 256, 0, stream>>>(tw, cb, addvec);

    const size_t xElems = (size_t)BDIM * KDIM;
    const size_t wElems = (size_t)KDIM * NDIM;
    const size_t need = 8192 + 2 * xElems * sizeof(unsigned short)
                             + 2 * wElems * sizeof(unsigned short);

    if (ws_size >= need) {
        unsigned short* xhi  = (unsigned short*)((char*)d_ws + 8192);
        unsigned short* xlo  = xhi + xElems;
        unsigned short* whit = xlo + xElems;
        unsigned short* wlot = whit + wElems;
        tnl_cvt_x_kernel<<<(int)(xElems / (256 * 8)), 256, 0, stream>>>(x, xhi, xlo);
        tnl_cvt_w_kernel<<<dim3(KDIM / 32, NDIM / 32), 256, 0, stream>>>(cw, whit, wlot);
        dim3 grid(NDIM / TN_A, BDIM / TM_A);  // (16, 32)
        tnl_gemm_async_kernel<<<grid, 256, 0, stream>>>(xhi, xlo, whit, wlot, addvec, out);
    } else {
        dim3 grid(NDIM / TN_F, BDIM / TM_F);  // (32, 32)
        tnl_gemm_fallback_kernel<<<grid, 256, 0, stream>>>(x, cw, addvec, out);
    }
}
